// MS_Attention_RepConv_qkv_id_79216376807551
// MI455X (gfx1250) — matrix operand order repaired and checked by reading, not compiled, not run on hardware
//
#include <hip/hip_runtime.h>

// MS_Attention_RepConv_qkv spiking attention for MI455X (gfx1250).
// All GEMMs via v_wmma_f32_16x16x32_f16 (wave32), spikes stored as f16 {0,1}.
// LIF recurrences fused into GEMM epilogues (block owns fixed tile, loops T).
// Block-shared GEMM operands (weights / kv tile) staged once in LDS with
// bank-conflict-free padded rows; spike operands stream from global (L2-resident).

#define T_  4
#define B_  16
#define C_  512
#define V_  1024
#define H_  8
#define DH_ 64

#define WLD  (C_ + 8)    // padded LDS row stride for 512-half weight rows
#define KVLD (DH_ + 8)   // padded LDS row stride for 64-half kv rows

typedef _Float16 h16;
typedef __attribute__((ext_vector_type(8)))  _Float16 v8h;
typedef __attribute__((ext_vector_type(16))) _Float16 v16h;
typedef __attribute__((ext_vector_type(8)))  float    v8f;

// ---- fragment helpers (per CDNA5 ISA 7.12.2 layouts, wave32) -------------
// A 16x32 f16, lane l: row = l&15, half = l>>4.
//   elements 0..7  hold K = half*8  + 0..7
//   elements 8..15 hold K = 16 + half*8 + 0..7
__device__ __forceinline__ v16h frag_a(const h16* base, int half) {
  const v8h lo = *(const v8h*)(base + half * 8);
  const v8h hi = *(const v8h*)(base + 16 + half * 8);
  v16h r;
#pragma unroll
  for (int i = 0; i < 8; ++i) { r[i] = lo[i]; r[i + 8] = hi[i]; }
  return r;
}
// B 32x16 f16, lane l: col = l&15, half = l>>4; same K order as frag_a
__device__ __forceinline__ v16h frag_b(const h16* base, int half) {
  const v8h lo = *(const v8h*)(base + half * 8);
  const v8h hi = *(const v8h*)(base + 16 + half * 8);
  v16h r;
#pragma unroll
  for (int i = 0; i < 8; ++i) { r[i] = lo[i]; r[i + 8] = hi[i]; }
  return r;
}
__device__ __forceinline__ v8f wmma16(v16h a, v16h b, v8f c) {
  return __builtin_amdgcn_wmma_f32_16x16x32_f16(false, a, false, b, (short)0, c,
                                                false, false);
}
__device__ __forceinline__ v8f zero8() {
  v8f z;
#pragma unroll
  for (int i = 0; i < 8; ++i) z[i] = 0.0f;
  return z;
}

// ---- weight f32 -> f16 pack: [Wq|Wk|Wv|Wp], each C*C -------------------
__global__ void cvt_w_kernel(const float* __restrict__ Wq, const float* __restrict__ Wk,
                             const float* __restrict__ Wv, const float* __restrict__ Wp,
                             h16* __restrict__ out) {
  int i = blockIdx.x * 256 + threadIdx.x;          // 0 .. 4*C*C-1
  int m = i >> 18;
  int off = i & ((1 << 18) - 1);
  const float* s = (m == 0) ? Wq : (m == 1) ? Wk : (m == 2) ? Wv : Wp;
  out[i] = (h16)s[off];
}

// ---- head LIF: x[t][b][c][v] f32 -> xsT[t][b][v][c] f16 (spikes) --------
__global__ void lif_head_kernel(const float* __restrict__ x, h16* __restrict__ xsT) {
  __shared__ h16 tile[64][72];                      // 64c x 64v, padded
  const int v0 = blockIdx.x * 64;
  const int c0 = blockIdx.y * 64;
  const int b  = blockIdx.z;
  const int tid = threadIdx.x;
  float vm[16];
#pragma unroll
  for (int i = 0; i < 16; ++i) vm[i] = 0.0f;
  for (int t = 0; t < T_; ++t) {
    const size_t xb = (size_t)(t * B_ + b) * C_ * V_;
#pragma unroll
    for (int i = 0; i < 16; ++i) {
      int e = i * 256 + tid;
      int cl = e >> 6, vl = e & 63;                 // coalesced along v
      float xv = x[xb + (size_t)(c0 + cl) * V_ + (v0 + vl)];
      float m  = 0.5f * (vm[i] + xv);               // v += (x - v)/tau, tau=2
      float s  = (m >= 1.0f) ? 1.0f : 0.0f;
      vm[i]    = (m >= 1.0f) ? 0.0f : m;            // hard reset to 0
      tile[cl][vl] = (h16)s;
    }
    __syncthreads();
    const size_t ob = (size_t)(t * B_ + b) * V_ * C_;
#pragma unroll
    for (int i = 0; i < 16; ++i) {
      int e = i * 256 + tid;
      int vl = e >> 6, cl = e & 63;                 // coalesced along c
      xsT[ob + (size_t)(v0 + vl) * C_ + (c0 + cl)] = tile[cl][vl];
    }
    __syncthreads();
  }
}

// cooperative copy of 64 weight rows [rowBase .. rowBase+63] into padded LDS
__device__ __forceinline__ void stage_w(h16* sW, const h16* __restrict__ W,
                                        int rowBase, int tid) {
#pragma unroll
  for (int i = 0; i < 16; ++i) {
    int u  = i * 256 + tid;                         // 4096 units of 8 halves
    int rw = u >> 6;                                // 0..63
    int ko = (u & 63) * 8;                          // 0..504
    *(v8h*)(sW + rw * WLD + ko) = *(const v8h*)(W + (size_t)(rowBase + rw) * C_ + ko);
  }
}

// ---- q GEMM + LIF:  M=v(1024), N=o(512), K=c(512); out qs[t][b][v][o] ----
// B operand (Wq rows for this block's 64 output channels) staged in LDS once.
__global__ void gemm_lif_q_kernel(const h16* __restrict__ xsT, const h16* __restrict__ W,
                                  const float* __restrict__ sc, const float* __restrict__ sh,
                                  h16* __restrict__ qs) {
  __shared__ h16 sW[64 * WLD];
  const int b  = blockIdx.z;
  const int M0 = blockIdx.x * 128;
  const int N0 = blockIdx.y * 64;
  const int lane = threadIdx.x & 31;
  const int wid  = threadIdx.x >> 5;
  const int wm = wid >> 1, wn = wid & 1;           // 4x2 waves, 32x32 per wave
  const int half = lane >> 4, l15 = lane & 15;

  stage_w(sW, W, N0, threadIdx.x);

  const int rowm[2] = { M0 + wm * 32 + l15, M0 + wm * 32 + 16 + l15 };
  const int coln[2] = { N0 + wn * 32 + l15, N0 + wn * 32 + 16 + l15 };
  const h16* bB0 = sW + (wn * 32 + l15) * WLD;
  const h16* bB1 = sW + (wn * 32 + 16 + l15) * WLD;
  float sN[2] = { sc[coln[0]], sc[coln[1]] };
  float bN[2] = { sh[coln[0]], sh[coln[1]] };

  float vm[2][2][8];
#pragma unroll
  for (int a = 0; a < 2; ++a)
#pragma unroll
    for (int c = 0; c < 2; ++c)
#pragma unroll
      for (int r = 0; r < 8; ++r) vm[a][c][r] = 0.0f;

  __syncthreads();

  for (int t = 0; t < T_; ++t) {
    const size_t tb = (size_t)(t * B_ + b);
    const h16* bA0 = xsT + (tb * V_ + rowm[0]) * C_;
    const h16* bA1 = xsT + (tb * V_ + rowm[1]) * C_;
    v8f acc[2][2];
    acc[0][0] = zero8(); acc[0][1] = zero8(); acc[1][0] = zero8(); acc[1][1] = zero8();
#pragma unroll 2
    for (int k = 0; k < C_; k += 32) {
      v16h a0 = frag_a(bA0 + k, half);
      v16h a1 = frag_a(bA1 + k, half);
      v16h b0 = frag_b(bB0 + k, half);
      v16h b1 = frag_b(bB1 + k, half);
      acc[0][0] = wmma16(a0, b0, acc[0][0]);
      acc[0][1] = wmma16(a0, b1, acc[0][1]);
      acc[1][0] = wmma16(a1, b0, acc[1][0]);
      acc[1][1] = wmma16(a1, b1, acc[1][1]);
    }
    const size_t ob = tb * V_ * C_;
#pragma unroll
    for (int mt = 0; mt < 2; ++mt)
#pragma unroll
      for (int nt = 0; nt < 2; ++nt)
#pragma unroll
        for (int r = 0; r < 8; ++r) {
          int m = M0 + wm * 32 + mt * 16 + r + 8 * half;   // v index
          int n = coln[nt];                                 // o index
          float y   = acc[mt][nt][r] * sN[nt] + bN[nt];
          float mem = 0.5f * (vm[mt][nt][r] + y);
          float s   = (mem >= 1.0f) ? 1.0f : 0.0f;
          vm[mt][nt][r] = (mem >= 1.0f) ? 0.0f : mem;
          qs[ob + (size_t)m * C_ + n] = (h16)s;
        }
  }
}

// ---- k/v GEMM + LIF: M=o(512), N=v(1024), K=c; out spk[t][b][o][v] -------
// A operand (weight rows) staged in LDS once per block.
__global__ void gemm_lif_kv_kernel(const h16* __restrict__ xsT, const h16* __restrict__ W,
                                   const float* __restrict__ sc, const float* __restrict__ sh,
                                   h16* __restrict__ spk) {
  __shared__ h16 sW[64 * WLD];
  const int b  = blockIdx.z;
  const int N0 = blockIdx.x * 128;
  const int M0 = blockIdx.y * 64;
  const int lane = threadIdx.x & 31;
  const int wid  = threadIdx.x >> 5;
  const int wm = wid >> 2, wn = wid & 3;           // 2x4 waves, 32x32 per wave
  const int half = lane >> 4, l15 = lane & 15;

  stage_w(sW, W, M0, threadIdx.x);

  const int coln[2] = { N0 + wn * 32 + l15, N0 + wn * 32 + 16 + l15 };
  const h16* bA0 = sW + (wm * 32 + l15) * WLD;
  const h16* bA1 = sW + (wm * 32 + 16 + l15) * WLD;

  float sM[2][8], bM[2][8];
#pragma unroll
  for (int mt = 0; mt < 2; ++mt)
#pragma unroll
    for (int r = 0; r < 8; ++r) {
      int m = M0 + wm * 32 + mt * 16 + r + 8 * half;
      sM[mt][r] = sc[m]; bM[mt][r] = sh[m];
    }

  float vm[2][2][8];
#pragma unroll
  for (int a = 0; a < 2; ++a)
#pragma unroll
    for (int c = 0; c < 2; ++c)
#pragma unroll
      for (int r = 0; r < 8; ++r) vm[a][c][r] = 0.0f;

  __syncthreads();

  for (int t = 0; t < T_; ++t) {
    const size_t tb = (size_t)(t * B_ + b);
    const h16* bB0 = xsT + (tb * V_ + coln[0]) * C_;
    const h16* bB1 = xsT + (tb * V_ + coln[1]) * C_;
    v8f acc[2][2];
    acc[0][0] = zero8(); acc[0][1] = zero8(); acc[1][0] = zero8(); acc[1][1] = zero8();
#pragma unroll 2
    for (int k = 0; k < C_; k += 32) {
      v16h a0 = frag_a(bA0 + k, half);
      v16h a1 = frag_a(bA1 + k, half);
      v16h b0 = frag_b(bB0 + k, half);
      v16h b1 = frag_b(bB1 + k, half);
      acc[0][0] = wmma16(a0, b0, acc[0][0]);
      acc[0][1] = wmma16(a0, b1, acc[0][1]);
      acc[1][0] = wmma16(a1, b0, acc[1][0]);
      acc[1][1] = wmma16(a1, b1, acc[1][1]);
    }
    const size_t ob = tb * C_ * V_;
#pragma unroll
    for (int mt = 0; mt < 2; ++mt)
#pragma unroll
      for (int nt = 0; nt < 2; ++nt)
#pragma unroll
        for (int r = 0; r < 8; ++r) {
          int m = M0 + wm * 32 + mt * 16 + r + 8 * half;   // o index
          int n = coln[nt];                                 // v index
          float y   = acc[mt][nt][r] * sM[mt][r] + bM[mt][r];
          float mem = 0.5f * (vm[mt][nt][r] + y);
          float s   = (mem >= 1.0f) ? 1.0f : 0.0f;
          vm[mt][nt][r] = (mem >= 1.0f) ? 0.0f : mem;
          spk[ob + (size_t)m * V_ + n] = (h16)s;
        }
  }
}

// ---- kvT[e][d] = sum_n v[e_g][n] * k[d_g][n]; M=e, N=d, K=n(1024) --------
__global__ void gemm_kvT_kernel(const h16* __restrict__ ks, const h16* __restrict__ vs,
                                h16* __restrict__ kvT) {
  const int gb = blockIdx.x;            // t*B*H
  const int h  = gb % H_;
  const int tb = gb / H_;
  const int lane = threadIdx.x & 31;
  const int wid  = threadIdx.x >> 5;
  const int wm = wid >> 2, wn = wid & 3;           // 2x4 waves, wave = 32x16
  const int half = lane >> 4, l15 = lane & 15;

  const h16* bA0 = vs + ((size_t)tb * C_ + h * DH_ + wm * 32 + l15) * V_;
  const h16* bA1 = vs + ((size_t)tb * C_ + h * DH_ + wm * 32 + 16 + l15) * V_;
  const h16* bB0 = ks + ((size_t)tb * C_ + h * DH_ + wn * 16 + l15) * V_;

  v8f acc[2];
  acc[0] = zero8(); acc[1] = zero8();
#pragma unroll 2
  for (int k = 0; k < V_; k += 32) {
    v16h a0 = frag_a(bA0 + k, half);
    v16h a1 = frag_a(bA1 + k, half);
    v16h b0 = frag_b(bB0 + k, half);
    acc[0] = wmma16(a0, b0, acc[0]);
    acc[1] = wmma16(a1, b0, acc[1]);
  }
  const size_t ob = ((size_t)tb * H_ + h) * DH_ * DH_;
#pragma unroll
  for (int mt = 0; mt < 2; ++mt)
#pragma unroll
    for (int r = 0; r < 8; ++r) {
      int e = wm * 32 + mt * 16 + r + 8 * half;
      int d = wn * 16 + l15;
      kvT[ob + (size_t)e * DH_ + d] = (h16)acc[mt][r];   // counts <=1024: f16 exact
    }
}

// ---- o = q @ kv * 0.125, then attn LIF (vth=0.5); out a[t][b][n][c] ------
// kvT tile (8KB) staged in LDS each timestep; shared by the whole block.
__global__ void gemm_o_lif_kernel(const h16* __restrict__ qs, const h16* __restrict__ kvT,
                                  h16* __restrict__ as_) {
  __shared__ h16 sKV[64 * KVLD];
  const int b  = blockIdx.z;
  const int h  = blockIdx.y;
  const int M0 = blockIdx.x * 128;                 // n tile
  const int lane = threadIdx.x & 31;
  const int wid  = threadIdx.x >> 5;
  const int wm = wid >> 1, wn = wid & 1;           // 4x2 waves, 32x32 per wave
  const int half = lane >> 4, l15 = lane & 15;

  const int rowm[2] = { M0 + wm * 32 + l15, M0 + wm * 32 + 16 + l15 };   // n
  const int coln[2] = { wn * 32 + l15, wn * 32 + 16 + l15 };             // e (0..63)
  const h16* bB0 = sKV + coln[0] * KVLD;
  const h16* bB1 = sKV + coln[1] * KVLD;

  float vm[2][2][8];
#pragma unroll
  for (int a = 0; a < 2; ++a)
#pragma unroll
    for (int c = 0; c < 2; ++c)
#pragma unroll
      for (int r = 0; r < 8; ++r) vm[a][c][r] = 0.0f;

  for (int t = 0; t < T_; ++t) {
    const size_t tb = (size_t)(t * B_ + b);
    const h16* kvbase = kvT + (tb * H_ + h) * DH_ * DH_;
    __syncthreads();                               // previous-iter readers done
#pragma unroll
    for (int i = 0; i < 2; ++i) {                  // 4096 halves = 512 v8h units
      int u  = i * 256 + threadIdx.x;
      int rw = u >> 3;                             // 0..63 (e row)
      int ko = (u & 7) * 8;                        // 0..56
      *(v8h*)(sKV + rw * KVLD + ko) = *(const v8h*)(kvbase + rw * DH_ + ko);
    }
    __syncthreads();

    const h16* bA0 = qs + (tb * V_ + rowm[0]) * C_ + h * DH_;
    const h16* bA1 = qs + (tb * V_ + rowm[1]) * C_ + h * DH_;
    v8f acc[2][2];
    acc[0][0] = zero8(); acc[0][1] = zero8(); acc[1][0] = zero8(); acc[1][1] = zero8();
#pragma unroll
    for (int k = 0; k < DH_; k += 32) {
      v16h a0 = frag_a(bA0 + k, half);
      v16h a1 = frag_a(bA1 + k, half);
      v16h b0 = frag_b(bB0 + k, half);
      v16h b1 = frag_b(bB1 + k, half);
      acc[0][0] = wmma16(a0, b0, acc[0][0]);
      acc[0][1] = wmma16(a0, b1, acc[0][1]);
      acc[1][0] = wmma16(a1, b0, acc[1][0]);
      acc[1][1] = wmma16(a1, b1, acc[1][1]);
    }
    const size_t ob = tb * V_ * C_;
#pragma unroll
    for (int mt = 0; mt < 2; ++mt)
#pragma unroll
      for (int nt = 0; nt < 2; ++nt)
#pragma unroll
        for (int r = 0; r < 8; ++r) {
          int m = M0 + wm * 32 + mt * 16 + r + 8 * half;   // n index
          int n = coln[nt] >= 0 ? (wn * 32 + nt * 16 + l15) : 0;  // e index
          float y   = acc[mt][nt][r] * 0.125f;              // attn_scale
          float mem = 0.5f * (vm[mt][nt][r] + y);
          float s   = (mem >= 0.5f) ? 1.0f : 0.0f;          // vth = 0.5
          vm[mt][nt][r] = (mem >= 0.5f) ? 0.0f : mem;
          as_[ob + (size_t)m * C_ + h * DH_ + n] = (h16)s;
        }
  }
}

// ---- final conv_bn: out[c_out][v] = (Wp @ a)*sp + bp ; f32 output --------
// A operand (Wp rows) staged in LDS once per block.
__global__ void gemm_p_kernel(const h16* __restrict__ as_, const h16* __restrict__ W,
                              const float* __restrict__ sc, const float* __restrict__ sh,
                              float* __restrict__ out) {
  __shared__ h16 sW[64 * WLD];
  const int tb = blockIdx.z;                        // t*B + b
  const int N0 = blockIdx.x * 128;                  // v tile
  const int M0 = blockIdx.y * 64;                   // c_out tile
  const int lane = threadIdx.x & 31;
  const int wid  = threadIdx.x >> 5;
  const int wm = wid >> 2, wn = wid & 3;            // 2x4 waves, 32x32 per wave
  const int half = lane >> 4, l15 = lane & 15;

  stage_w(sW, W, M0, threadIdx.x);

  const int coln[2] = { N0 + wn * 32 + l15, N0 + wn * 32 + 16 + l15 };
  const h16* bA0 = sW + (wm * 32 + l15) * WLD;
  const h16* bA1 = sW + (wm * 32 + 16 + l15) * WLD;
  const h16* bB0 = as_ + ((size_t)tb * V_ + coln[0]) * C_;
  const h16* bB1 = as_ + ((size_t)tb * V_ + coln[1]) * C_;

  float sM[2][8], bM[2][8];
#pragma unroll
  for (int mt = 0; mt < 2; ++mt)
#pragma unroll
    for (int r = 0; r < 8; ++r) {
      int m = M0 + wm * 32 + mt * 16 + r + 8 * half;
      sM[mt][r] = sc[m]; bM[mt][r] = sh[m];
    }

  __syncthreads();

  v8f acc[2][2];
  acc[0][0] = zero8(); acc[0][1] = zero8(); acc[1][0] = zero8(); acc[1][1] = zero8();
#pragma unroll 2
  for (int k = 0; k < C_; k += 32) {
    v16h a0 = frag_a(bA0 + k, half);
    v16h a1 = frag_a(bA1 + k, half);
    v16h b0 = frag_b(bB0 + k, half);
    v16h b1 = frag_b(bB1 + k, half);
    acc[0][0] = wmma16(a0, b0, acc[0][0]);
    acc[0][1] = wmma16(a0, b1, acc[0][1]);
    acc[1][0] = wmma16(a1, b0, acc[1][0]);
    acc[1][1] = wmma16(a1, b1, acc[1][1]);
  }
  const size_t ob = (size_t)tb * C_ * V_;
#pragma unroll
  for (int mt = 0; mt < 2; ++mt)
#pragma unroll
    for (int nt = 0; nt < 2; ++nt)
#pragma unroll
      for (int r = 0; r < 8; ++r) {
        int m = M0 + wm * 32 + mt * 16 + r + 8 * half;     // c_out
        int n = coln[nt];                                   // v (coalesced f32)
        out[ob + (size_t)m * V_ + n] = acc[mt][nt][r] * sM[mt][r] + bM[mt][r];
      }
}

extern "C" void kernel_launch(void* const* d_in, const int* in_sizes, int n_in,
                              void* d_out, int out_size, void* d_ws, size_t ws_size,
                              hipStream_t stream) {
  (void)in_sizes; (void)n_in; (void)out_size; (void)ws_size;
  const float* x  = (const float*)d_in[0];
  const float* Wq = (const float*)d_in[1];
  const float* sq = (const float*)d_in[2];
  const float* bq = (const float*)d_in[3];
  const float* Wk = (const float*)d_in[4];
  const float* sk = (const float*)d_in[5];
  const float* bk = (const float*)d_in[6];
  const float* Wv = (const float*)d_in[7];
  const float* sv = (const float*)d_in[8];
  const float* bv = (const float*)d_in[9];
  const float* Wp = (const float*)d_in[10];
  const float* sp = (const float*)d_in[11];
  const float* bp = (const float*)d_in[12];
  float* out = (float*)d_out;

  char* ws = (char*)d_ws;
  const size_t SPK = (size_t)T_ * B_ * C_ * V_ * sizeof(h16);   // 64 MB per spike tensor
  h16* xsT = (h16*)(ws);                                        // [t][b][v][c]
  h16* qs  = (h16*)(ws + SPK);                                  // [t][b][v][c]
  h16* ks  = (h16*)(ws + 2 * SPK);                              // [t][b][c][v]
  h16* vs  = (h16*)(ws + 3 * SPK);                              // [t][b][c][v]
  h16* as_ = (h16*)(ws + 4 * SPK);                              // [t][b][v][c]
  h16* kvT = (h16*)(ws + 5 * SPK);                              // [t][b][h][e][d]
  h16* Wh  = (h16*)(ws + 5 * SPK + (size_t)T_ * B_ * H_ * DH_ * DH_ * sizeof(h16));
  h16* Wqh = Wh;
  h16* Wkh = Wh + (size_t)C_ * C_;
  h16* Wvh = Wh + 2 * (size_t)C_ * C_;
  h16* Wph = Wh + 3 * (size_t)C_ * C_;

  cvt_w_kernel<<<dim3(4096), 256, 0, stream>>>(Wq, Wk, Wv, Wp, Wh);
  lif_head_kernel<<<dim3(V_ / 64, C_ / 64, B_), 256, 0, stream>>>(x, xsT);
  gemm_lif_q_kernel <<<dim3(V_ / 128, C_ / 64, B_), 256, 0, stream>>>(xsT, Wqh, sq, bq, qs);
  gemm_lif_kv_kernel<<<dim3(V_ / 128, C_ / 64, B_), 256, 0, stream>>>(xsT, Wkh, sk, bk, ks);
  gemm_lif_kv_kernel<<<dim3(V_ / 128, C_ / 64, B_), 256, 0, stream>>>(xsT, Wvh, sv, bv, vs);
  gemm_kvT_kernel<<<dim3(T_ * B_ * H_), 256, 0, stream>>>(ks, vs, kvT);
  gemm_o_lif_kernel<<<dim3(V_ / 128, H_, B_), 256, 0, stream>>>(qs, kvT, as_);
  gemm_p_kernel<<<dim3(V_ / 128, C_ / 64, T_ * B_), 256, 0, stream>>>(as_, Wph, sp, bp, out);
}
